// F0EnergyEncoder_8005819039776
// MI455X (gfx1250) — matrix operand order repaired and checked by reading, not compiled, not run on hardware
//
#include <hip/hip_runtime.h>
#include <hip/hip_bf16.h>
#include <math.h>

typedef __bf16 bf16;
typedef __attribute__((ext_vector_type(16))) __bf16 v16bf;
typedef __attribute__((ext_vector_type(8)))  __bf16 v8bf;
typedef __attribute__((ext_vector_type(8)))  float   v8f;

#define NFFT     2048
#define WINLEN   960
#define HOPSZ    240
#define TFRAMES  1001
#define BATCH    16
#define DPAD     320      // 301 padded to multiple of 32
#define HID      128
#define G4       512      // 4*HID
#define OUTD     64
#define LENX     240000
#define NROWS    (TFRAMES * BATCH)   // 16016

__device__ __forceinline__ float sigf(float x) { return 1.0f / (1.0f + __expf(-x)); }

// ---------------------------------------------------------------------------
// Weight convert (f32 -> bf16) with optional K padding to kpad
// ---------------------------------------------------------------------------
__global__ void k_cvt_pad(const float* __restrict__ src, bf16* __restrict__ dst,
                          int rows, int kin, int kpad) {
  int i = blockIdx.x * 256 + threadIdx.x;
  if (i >= rows * kpad) return;
  int g = i / kpad, k = i % kpad;
  dst[i] = (bf16)((k < kin) ? src[g * kin + k] : 0.0f);
}

// ---------------------------------------------------------------------------
// Fused STFT power + harmonic feature extraction. One block per (t, b).
// Radix-2 DIT FFT entirely in LDS; power spectrum never touches HBM.
// ---------------------------------------------------------------------------
__global__ __launch_bounds__(256)
void k_stft_feat(const float* __restrict__ x, const float* __restrict__ f0,
                 bf16* __restrict__ feat) {
  __shared__ float re[NFFT];
  __shared__ float im[NFFT];
  const int t = blockIdx.x;
  const int b = blockIdx.y;
  const int tid = threadIdx.x;
  const float* xb = x + (size_t)b * LENX;

  // Windowed frame, loaded in bit-reversed order. Hann window occupies
  // positions [544, 1504); everything else is zero. Reflect padding of 1024.
  for (int p = tid; p < NFFT; p += 256) {
    float v = 0.0f;
    if (p >= 544 && p < 544 + WINLEN) {
      int s = p - 544;
      int j = t * HOPSZ + s - 480;
      if (j < 0) j = -j;
      if (j >= LENX) j = 2 * LENX - 2 - j;
      float w = 0.5f - 0.5f * __cosf(6.28318530718f * (float)s / (float)WINLEN);
      v = xb[j] * w;
    }
    int r = (int)(__brev((unsigned)p) >> 21);   // 11-bit reversal
    re[r] = v;
    im[r] = 0.0f;
  }
  __syncthreads();

  // 11 radix-2 stages, 1024 butterflies/stage, 4 per thread.
  for (int s = 1; s <= 11; ++s) {
    const int half = 1 << (s - 1);
    const float ang = -6.28318530718f / (float)(1 << s);
    for (int bf = tid; bf < NFFT / 2; bf += 256) {
      int pos = bf & (half - 1);
      int i0 = ((bf >> (s - 1)) << s) + pos;
      int i1 = i0 + half;
      float sn, cs;
      __sincosf(ang * (float)pos, &sn, &cs);
      float ur = re[i0], ui = im[i0];
      float vr = re[i1], vi = im[i1];
      float tr = vr * cs - vi * sn;
      float ti = vr * sn + vi * cs;
      re[i0] = ur + tr; im[i0] = ui + ti;
      re[i1] = ur - tr; im[i1] = ui - ti;
    }
    __syncthreads();
  }

  // Power spectrum into re[0..1024]; reference forces last bin to 0.
  for (int k = tid; k <= 1024; k += 256) {
    float p = re[k] * re[k] + im[k] * im[k];
    re[k] = (k == 1024) ? 0.0f : p;
  }
  __syncthreads();

  // Harmonic gather straight from LDS.
  const float f0v = f0[(size_t)b * TFRAMES + t];
  const float f0nz = (f0v > 0.0f) ? f0v : 80.0f;     // SR/NUM_BANDS*0.5
  const float inv_fi = 2048.0f / 24000.0f;           // 1/FREQ_INTERVAL
  const float inv_rng = 1.0f / 23.0f;                // 1/(LOG_E_MAX-LOG_E_MIN)
  const size_t rowo = ((size_t)t * BATCH + b) * DPAD;
  for (int d = tid; d < DPAD; d += 256) {
    float outv;
    if (d < 300) {
      float mult = (d == 0) ? 0.5f : (1.0f + 0.5f * (float)(d - 1));
      int idx = __float2int_rn(f0nz * mult * inv_fi);
      idx = (idx < 0) ? 0 : ((idx > 1024) ? 1024 : idx);
      outv = (__logf(re[idx] + 1e-8f) + 18.0f) * inv_rng;
    } else if (d == 300) {
      outv = __logf(f0nz);
    } else {
      outv = 0.0f;                                   // K padding
    }
    feat[rowo + d] = (bf16)outv;
  }
}

// ---------------------------------------------------------------------------
// Input projection: pre[dir] = feat(16016 x 320) @ w_ih^T(320 x 512) + b
// One WMMA 16x16 output tile per wave; K loop of 10 x bf16-K32 steps.
// ---------------------------------------------------------------------------
__global__ __launch_bounds__(256)
void k_inproj(const bf16* __restrict__ feat,
              const bf16* __restrict__ wih_f, const bf16* __restrict__ wih_b,
              const float* __restrict__ bias_f, const float* __restrict__ bias_b,
              float* __restrict__ pre_f, float* __restrict__ pre_b) {
  const int lane = threadIdx.x & 31;
  const int wave = threadIdx.x >> 5;
  const int mtile = blockIdx.x;
  const int ntile = blockIdx.y * 8 + wave;
  const int dir = blockIdx.z;
  const bf16* w = dir ? wih_b : wih_f;
  const float* bias = dir ? bias_b : bias_f;
  float* pre = dir ? pre_b : pre_f;

  const int hi = (lane >= 16) ? 1 : 0;
  const int mrow = mtile * 16 + (lane & 15);
  const int ncol = ntile * 16 + (lane & 15);
  const bf16* arow = feat + (size_t)mrow * DPAD + hi * 16;
  const bf16* brow = w + (size_t)ncol * DPAD + hi * 16;

  v8f acc = {};
  for (int kt = 0; kt < DPAD / 32; ++kt) {
    v8bf alo = *(const v8bf*)(arow + kt * 32);
    v8bf ahi = *(const v8bf*)(arow + kt * 32 + 8);
    v16bf a;
#pragma unroll
    for (int i = 0; i < 8; ++i) { a[i] = alo[i]; a[8 + i] = ahi[i]; }
    v16bf bm = *(const v16bf*)(brow + kt * 32);
    acc = __builtin_amdgcn_wmma_f32_16x16x32_bf16(false, a, false, bm,
                                                  (short)0, acc, false, false);
  }
  const float bv = bias[ncol];
#pragma unroll
  for (int r = 0; r < 8; ++r) {
    int m = mtile * 16 + r + hi * 8;
    pre[(size_t)m * G4 + ncol] = acc[r] + bv;
  }
}

// ---------------------------------------------------------------------------
// Persistent LSTM scan. One block per direction; 16 waves cover the 32
// N-tiles of the 16x512 recurrent GEMM each step. h/c live in LDS.
// ---------------------------------------------------------------------------
__global__ __launch_bounds__(512)
void k_scan(const float* __restrict__ pre_f, const float* __restrict__ pre_b,
            const bf16* __restrict__ whh_f, const bf16* __restrict__ whh_b,
            float* __restrict__ hf, float* __restrict__ hb) {
  __shared__ float z[BATCH * G4];     // gate pre-activations 16x512
  __shared__ float cst[BATCH * HID];  // cell state
  __shared__ bf16  hst[BATCH * HID];  // hidden state (bf16 -> WMMA A operand)

  const int dir = blockIdx.x;
  const float* pre = dir ? pre_b : pre_f;
  const bf16* whh = dir ? whh_b : whh_f;
  float* hout = dir ? hb : hf;

  const int tid = threadIdx.x;
  const int lane = tid & 31;
  const int wave = tid >> 5;
  const int hi = (lane >= 16) ? 1 : 0;

  for (int e = tid; e < BATCH * HID; e += 512) { cst[e] = 0.0f; hst[e] = (bf16)0.0f; }
  __syncthreads();

  for (int step = 0; step < TFRAMES; ++step) {
    const int t = dir ? (TFRAMES - 1 - step) : step;
    const float* prow = pre + (size_t)t * BATCH * G4;
    if (step + 1 < TFRAMES) {   // prefetch next step's gate pre-activations
      const int tn = dir ? (TFRAMES - 2 - step) : (step + 1);
      __builtin_prefetch(pre + (size_t)tn * BATCH * G4 + tid * 16, 0, 1);
    }

    // GEMM phase: z = pre[t] + h @ w_hh^T  (each wave: 2 N-tiles, K=128)
    v8f acc[2];
#pragma unroll
    for (int q = 0; q < 2; ++q) {
      const int nt = wave * 2 + q;
      const int ncol = nt * 16 + (lane & 15);
      v8f a0 = {};
#pragma unroll
      for (int r = 0; r < 8; ++r)
        a0[r] = prow[(size_t)(r + hi * 8) * G4 + ncol];
      const bf16* brow = whh + (size_t)ncol * HID + hi * 16;
#pragma unroll
      for (int kt = 0; kt < HID / 32; ++kt) {
        v16bf am;
        const int m = lane & 15;
        const int kb = kt * 32 + hi * 16;
#pragma unroll
        for (int i = 0; i < 8; ++i) {
          am[i]     = hst[m * HID + kb + i];
          am[8 + i] = hst[m * HID + kb + 8 + i];
        }
        v16bf bm = *(const v16bf*)(brow + kt * 32);
        a0 = __builtin_amdgcn_wmma_f32_16x16x32_bf16(false, am, false, bm,
                                                     (short)0, a0, false, false);
      }
      acc[q] = a0;
    }
#pragma unroll
    for (int q = 0; q < 2; ++q) {
      const int nt = wave * 2 + q;
      const int ncol = nt * 16 + (lane & 15);
#pragma unroll
      for (int r = 0; r < 8; ++r)
        z[(r + hi * 8) * G4 + ncol] = acc[q][r];
    }
    __syncthreads();

    // Gate phase: c,h update; write h back to LDS (bf16) and global (f32)
    for (int e = tid; e < BATCH * HID; e += 512) {
      int bb = e >> 7, j = e & 127;
      float iv = z[bb * G4 + j];
      float fv = z[bb * G4 + HID + j];
      float gv = z[bb * G4 + 2 * HID + j];
      float ov = z[bb * G4 + 3 * HID + j];
      float c = sigf(fv) * cst[e] + sigf(iv) * tanhf(gv);
      cst[e] = c;
      float h = sigf(ov) * tanhf(c);
      hst[e] = (bf16)h;
      hout[((size_t)t * BATCH + bb) * HID + j] = h;
    }
    __syncthreads();
  }
}

// ---------------------------------------------------------------------------
// LayerNorm over concat(hf, hb) (256 per row). One wave per row.
// ---------------------------------------------------------------------------
__global__ __launch_bounds__(256)
void k_layernorm(const float* __restrict__ hf, const float* __restrict__ hb,
                 const float* __restrict__ ln_g, const float* __restrict__ ln_b,
                 bf16* __restrict__ hn) {
  const int lane = threadIdx.x & 31;
  const int wave = threadIdx.x >> 5;
  const int row = blockIdx.x * 8 + wave;
  if (row >= NROWS) return;
  float v[8], s = 0.f, s2 = 0.f;
#pragma unroll
  for (int q = 0; q < 8; ++q) {
    int j = lane + 32 * q;
    float xv = (j < HID) ? hf[(size_t)row * HID + j]
                         : hb[(size_t)row * HID + (j - HID)];
    v[q] = xv; s += xv; s2 += xv * xv;
  }
#pragma unroll
  for (int m = 16; m >= 1; m >>= 1) {
    s += __shfl_xor(s, m, 32);
    s2 += __shfl_xor(s2, m, 32);
  }
  float mu = s * (1.0f / 256.0f);
  float var = s2 * (1.0f / 256.0f) - mu * mu;
  float rstd = rsqrtf(var + 1e-5f);
#pragma unroll
  for (int q = 0; q < 8; ++q) {
    int j = lane + 32 * q;
    hn[(size_t)row * 256 + j] = (bf16)((v[q] - mu) * rstd * ln_g[j] + ln_b[j]);
  }
}

// ---------------------------------------------------------------------------
// Output projection: out = hn(16016 x 256) @ out_w^T(256 x 64) + out_b
// 4 waves per block, one 16x16 tile each; K loop of 8 bf16-K32 steps.
// ---------------------------------------------------------------------------
__global__ __launch_bounds__(128)
void k_outproj(const bf16* __restrict__ hn, const bf16* __restrict__ outw,
               const float* __restrict__ outb, float* __restrict__ out) {
  const int lane = threadIdx.x & 31;
  const int nt = threadIdx.x >> 5;  // 0..3
  const int mtile = blockIdx.x;
  const int hi = (lane >= 16) ? 1 : 0;
  const int mrow = mtile * 16 + (lane & 15);
  const int ncol = nt * 16 + (lane & 15);
  const bf16* arow = hn + (size_t)mrow * 256 + hi * 16;
  const bf16* brow = outw + (size_t)ncol * 256 + hi * 16;

  v8f acc = {};
#pragma unroll
  for (int kt = 0; kt < 8; ++kt) {
    v8bf alo = *(const v8bf*)(arow + kt * 32);
    v8bf ahi = *(const v8bf*)(arow + kt * 32 + 8);
    v16bf a;
#pragma unroll
    for (int i = 0; i < 8; ++i) { a[i] = alo[i]; a[8 + i] = ahi[i]; }
    v16bf bm = *(const v16bf*)(brow + kt * 32);
    acc = __builtin_amdgcn_wmma_f32_16x16x32_bf16(false, a, false, bm,
                                                  (short)0, acc, false, false);
  }
  const float bv = outb[ncol];
#pragma unroll
  for (int r = 0; r < 8; ++r) {
    int m = mtile * 16 + r + hi * 8;
    int t = m >> 4, bb = m & 15;
    out[((size_t)bb * TFRAMES + t) * OUTD + ncol] = acc[r] + bv;
  }
}

// ---------------------------------------------------------------------------
extern "C" void kernel_launch(void* const* d_in, const int* in_sizes, int n_in,
                              void* d_out, int out_size, void* d_ws, size_t ws_size,
                              hipStream_t stream) {
  (void)in_sizes; (void)n_in; (void)out_size; (void)ws_size;
  const float* x     = (const float*)d_in[0];
  const float* f0    = (const float*)d_in[1];
  const float* wih_f = (const float*)d_in[2];
  const float* whh_f = (const float*)d_in[3];
  const float* b_f   = (const float*)d_in[4];
  const float* wih_b = (const float*)d_in[5];
  const float* whh_b = (const float*)d_in[6];
  const float* b_b   = (const float*)d_in[7];
  const float* ln_g  = (const float*)d_in[8];
  const float* ln_b  = (const float*)d_in[9];
  const float* outw  = (const float*)d_in[10];
  const float* outb  = (const float*)d_in[11];
  float* out = (float*)d_out;

  char* ws = (char*)d_ws;
  size_t off = 0;
  auto take = [&](size_t bytes) -> char* {
    char* p = ws + off;
    off += (bytes + 255) & ~(size_t)255;
    return p;
  };
  bf16*  feat   = (bf16*) take((size_t)NROWS * DPAD * 2);
  float* pre_f  = (float*)take((size_t)NROWS * G4 * 4);
  float* pre_b  = (float*)take((size_t)NROWS * G4 * 4);
  float* hfb    = (float*)take((size_t)NROWS * HID * 4);
  float* hbb    = (float*)take((size_t)NROWS * HID * 4);
  bf16*  hn     = (bf16*) take((size_t)NROWS * 256 * 2);
  bf16*  wihf16 = (bf16*) take((size_t)G4 * DPAD * 2);
  bf16*  wihb16 = (bf16*) take((size_t)G4 * DPAD * 2);
  bf16*  whhf16 = (bf16*) take((size_t)G4 * HID * 2);
  bf16*  whhb16 = (bf16*) take((size_t)G4 * HID * 2);
  bf16*  outw16 = (bf16*) take((size_t)OUTD * 256 * 2);

  k_cvt_pad<<<(G4 * DPAD + 255) / 256, 256, 0, stream>>>(wih_f, wihf16, G4, 301, DPAD);
  k_cvt_pad<<<(G4 * DPAD + 255) / 256, 256, 0, stream>>>(wih_b, wihb16, G4, 301, DPAD);
  k_cvt_pad<<<(G4 * HID + 255) / 256, 256, 0, stream>>>(whh_f, whhf16, G4, HID, HID);
  k_cvt_pad<<<(G4 * HID + 255) / 256, 256, 0, stream>>>(whh_b, whhb16, G4, HID, HID);
  k_cvt_pad<<<(OUTD * 256 + 255) / 256, 256, 0, stream>>>(outw, outw16, OUTD, 256, 256);

  k_stft_feat<<<dim3(TFRAMES, BATCH), 256, 0, stream>>>(x, f0, feat);
  k_inproj<<<dim3(TFRAMES, 4, 2), 256, 0, stream>>>(feat, wihf16, wihb16,
                                                    b_f, b_b, pre_f, pre_b);
  k_scan<<<2, 512, 0, stream>>>(pre_f, pre_b, whhf16, whhb16, hfb, hbb);
  k_layernorm<<<(NROWS / 8), 256, 0, stream>>>(hfb, hbb, ln_g, ln_b, hn);
  k_outproj<<<TFRAMES, 128, 0, stream>>>(hn, outw16, outb, out);
}
